// GAT_51161650430432
// MI455X (gfx1250) — matrix operand order repaired and checked by reading, not compiled, not run on hardware
//
#include <hip/hip_runtime.h>
#include <math.h>

typedef __attribute__((ext_vector_type(16))) __bf16 v16bf;
typedef __attribute__((ext_vector_type(8)))  float v8f;
typedef __attribute__((ext_vector_type(4)))  unsigned short ush4;

// ---------- helpers ----------
__device__ inline unsigned short f2bf(float f) {
  unsigned int u = __float_as_uint(f);
  u += 0x7FFFu + ((u >> 16) & 1u);   // round-to-nearest-even
  return (unsigned short)(u >> 16);
}
__device__ inline void atomicMaxF(float* addr, float val) {
  // standard ordered-int trick; init value must be -inf
  if (val >= 0.0f) atomicMax((int*)addr, __float_as_int(val));
  else             atomicMin((unsigned int*)addr, (unsigned int)__float_as_int(val));
}

struct Frag32B { uint4 lo, hi; };   // 32 bytes -> bit_cast to v16bf

// ---------- bf16-WMMA GEMM: Y[N,M] = X[N,K] @ W[K,M] (+bias) ----------
// Compile-time K,M. block = 128 threads (4 wave32), 64 rows/block.
// LDS: Xs[64][K] bf16 row-major; WsT[M][K] bf16 (W transposed) so each lane's
// A and B fragments are contiguous 16B chunks -> ds_load_b128.
template <int K, int M>
__global__ __launch_bounds__(128) void gemm_bf16_wmma(
    const float* __restrict__ X, const float* __restrict__ W,
    const float* __restrict__ bias, float* __restrict__ Y, int N)
{
  __shared__ unsigned short Xs[64 * K];
  __shared__ unsigned short WsT[M * K];
  const int tid = threadIdx.x;
  const int rowBase = blockIdx.x * 64;
  const bool fullBlock = (rowBase + 64 <= N);

  // W: coalesced global read, transposed scatter into LDS (done once per block)
  for (int i = tid; i < K * M; i += 128) {
    int k = i / M, m = i - k * M;
    WsT[m * K + k] = f2bf(W[i]);
  }
  // X tile: float4 coalesced reads -> ush4 LDS stores
  for (int i = tid; i < 64 * K / 4; i += 128) {
    int r = (i * 4) / K, c = (i * 4) - r * K;
    float4 v = make_float4(0.f, 0.f, 0.f, 0.f);
    if (fullBlock || rowBase + r < N) v = *(const float4*)(X + (long long)(rowBase + r) * K + c);
    ush4 o; o.x = f2bf(v.x); o.y = f2bf(v.y); o.z = f2bf(v.z); o.w = f2bf(v.w);
    *(ush4*)(Xs + i * 4) = o;
  }
  __syncthreads();

  const int wave = tid >> 5;
  const int lane = tid & 31;
  const int half = lane >> 4;      // lane half (0/1)
  const int l16  = lane & 15;
  const unsigned short* ax = Xs + (wave * 16 + l16) * K;   // A: row = lane%16

#pragma unroll
  for (int mt = 0; mt < M; mt += 16) {
    v8f acc = {};
    const unsigned short* bx = WsT + (mt + l16) * K;        // B: col = lane%16
#pragma unroll
    for (int k0 = 0; k0 < K; k0 += 32) {
      Frag32B at, bt;
      // A 16x32 bf16: per lane two contiguous 8-elem chunks at k0+8*half and k0+16+8*half
      at.lo = *(const uint4*)(ax + k0 + half * 8);
      at.hi = *(const uint4*)(ax + k0 + 16 + half * 8);
      // B 32x16 bf16 (K-major in LDS): two contiguous 8-elem chunks at k0+8*half and k0+16+8*half
      bt.lo = *(const uint4*)(bx + k0 + half * 8);
      bt.hi = *(const uint4*)(bx + k0 + 16 + half * 8);
      v16bf a = __builtin_bit_cast(v16bf, at);
      v16bf b = __builtin_bit_cast(v16bf, bt);
      acc = __builtin_amdgcn_wmma_f32_16x16x32_bf16(
          false, a, false, b, (short)0, acc, false, false);
    }
    const float bv = bias ? bias[mt + l16] : 0.0f;
    const int r0 = rowBase + wave * 16 + half * 8;          // D: row = 8*half+v, col = lane%16
    float* yp = Y + (long long)r0 * M + mt + l16;
    if (fullBlock) {
#pragma unroll
      for (int vv = 0; vv < 8; ++vv) yp[(long long)vv * M] = acc[vv] + bv;
    } else {
#pragma unroll
      for (int vv = 0; vv < 8; ++vv)
        if (r0 + vv < N) yp[(long long)vv * M] = acc[vv] + bv;
    }
  }
}

// ---------- elementwise / score / GAT kernels ----------
__global__ void fill_kernel(float* p, long long n, float v) {
  long long i = (long long)blockIdx.x * blockDim.x + threadIdx.x;
  if (i < n) p[i] = v;
}

__global__ void elu_kernel(const float* __restrict__ in, float* __restrict__ out, long long n) {
  long long i = (long long)blockIdx.x * blockDim.x + threadIdx.x;
  if (i < n) { float x = in[i]; out[i] = (x > 0.0f) ? x : (__expf(x) - 1.0f); }
}

// out[n*H+h] = sum_f feat[n,h,f] * avec[h,f]
__global__ void scores_kernel(const float* __restrict__ feat, const float* __restrict__ avec,
                              float* __restrict__ out, int N, int H, int F) {
  int i = blockIdx.x * blockDim.x + threadIdx.x;
  if (i >= N * H) return;
  int n = i / H, h = i - n * H;
  const float* fp = feat + (long long)n * H * F + h * F;
  const float* ap = avec + h * F;
  float s = 0.0f;
  for (int f = 0; f < F; ++f) s += fp[f] * ap[f];
  out[i] = s;
}

__global__ void edge_max_kernel(const int* __restrict__ src, const int* __restrict__ dst,
                                const float* __restrict__ ssrc, const float* __restrict__ sdst,
                                float* __restrict__ m, int E, int Nself, int H) {
  int total = (E + Nself) * H;
  int i = blockIdx.x * blockDim.x + threadIdx.x;
  if (i >= total) return;
  int e = i / H, h = i - e * H;
  int s, d;
  if (e < E) { s = src[e]; d = dst[e]; } else { s = d = e - E; }
  float x = ssrc[s * H + h] + sdst[d * H + h];
  x = (x > 0.0f) ? x : 0.2f * x;                 // leaky_relu 0.2
  atomicMaxF(&m[d * H + h], x);
}

__global__ void edge_expsum_kernel(const int* __restrict__ src, const int* __restrict__ dst,
                                   const float* __restrict__ ssrc, const float* __restrict__ sdst,
                                   const float* __restrict__ m, float* __restrict__ den,
                                   float* __restrict__ exb, int E, int Nself, int H) {
  int total = (E + Nself) * H;
  int i = blockIdx.x * blockDim.x + threadIdx.x;
  if (i >= total) return;
  int e = i / H, h = i - e * H;
  int s, d;
  if (e < E) { s = src[e]; d = dst[e]; } else { s = d = e - E; }
  float x = ssrc[s * H + h] + sdst[d * H + h];
  x = (x > 0.0f) ? x : 0.2f * x;
  float ex = __expf(x - m[d * H + h]);
  exb[i] = ex;
  atomicAdd(&den[d * H + h], ex);
}

__global__ void edge_scatter_kernel(const int* __restrict__ src, const int* __restrict__ dst,
                                    const float* __restrict__ feat, const float* __restrict__ exb,
                                    const float* __restrict__ den, float* __restrict__ out,
                                    int E, int Nself, int H, int F, int ostride, int cofs) {
  const int hf = H * F;
  long long total = (long long)(E + Nself) * hf;
  long long i = (long long)blockIdx.x * blockDim.x + threadIdx.x;
  if (i >= total) return;
  int e = (int)(i / hf);
  int r = (int)(i - (long long)e * hf);
  int h = r / F;
  int s, d;
  if (e < E) { s = src[e]; d = dst[e]; } else { s = d = e - E; }
  float alpha = exb[(long long)e * H + h] / den[d * H + h];
  atomicAdd(&out[(long long)d * ostride + cofs + r],
            feat[(long long)s * hf + r] * alpha);
}

__global__ void biasact_kernel(float* __restrict__ Y, const float* __restrict__ b,
                               int N, int C, int stride, int cofs, int relu) {
  long long i = (long long)blockIdx.x * blockDim.x + threadIdx.x;
  if (i >= (long long)N * C) return;
  int n = (int)(i / C), c = (int)(i - (long long)n * C);
  float* p = Y + (long long)n * stride + cofs + c;
  float v = *p + b[c];
  if (relu) v = fmaxf(v, 0.0f);
  *p = v;
}

// ---------- SAG pooling ----------
__global__ void sag_agg_kernel(const int* __restrict__ src, const int* __restrict__ dst,
                               const float* __restrict__ rep, float* __restrict__ agg, int E) {
  long long total = (long long)E * 128;
  long long i = (long long)blockIdx.x * blockDim.x + threadIdx.x;
  if (i >= total) return;
  int e = (int)(i >> 7), f = (int)(i & 127);
  atomicAdd(&agg[(long long)dst[e] * 128 + f], rep[(long long)src[e] * 128 + f]);
}

__global__ void sag_score_kernel(const float* __restrict__ rep, const float* __restrict__ agg,
                                 const float* __restrict__ Wr, const float* __restrict__ Wn,
                                 const float* __restrict__ b, float* __restrict__ sc, int N) {
  int n = blockIdx.x * blockDim.x + threadIdx.x;
  if (n >= N) return;
  float s = b[0];
  const float* rp = rep + (long long)n * 128;
  const float* ap = agg + (long long)n * 128;
  for (int f = 0; f < 128; ++f) s += rp[f] * Wr[f] + ap[f] * Wn[f];
  sc[n] = s;
}

__global__ void seg_max_kernel(const float* __restrict__ sc, const int* __restrict__ batch,
                               float* __restrict__ m, int N) {
  int n = blockIdx.x * blockDim.x + threadIdx.x;
  if (n < N) atomicMaxF(&m[batch[n]], sc[n]);
}

__global__ void seg_expsum_kernel(const float* __restrict__ sc, const int* __restrict__ batch,
                                  const float* __restrict__ m, float* __restrict__ den,
                                  float* __restrict__ ex, int N) {
  int n = blockIdx.x * blockDim.x + threadIdx.x;
  if (n >= N) return;
  float e = __expf(sc[n] - m[batch[n]]);
  ex[n] = e;
  atomicAdd(&den[batch[n]], e);
}

__global__ void sag_pool_kernel(const float* __restrict__ rep, const int* __restrict__ batch,
                                const float* __restrict__ ex, const float* __restrict__ den,
                                float* __restrict__ emb, int N) {
  long long total = (long long)N * 128;
  long long i = (long long)blockIdx.x * blockDim.x + threadIdx.x;
  if (i >= total) return;
  int n = (int)(i >> 7), f = (int)(i & 127);
  int g = batch[n];
  float sm = ex[n] / den[g];
  atomicAdd(&emb[(long long)g * 128 + f], rep[(long long)n * 128 + f] * sm);
}

// ---------- driver ----------
extern "C" void kernel_launch(void* const* d_in, const int* in_sizes, int n_in,
                              void* d_out, int out_size, void* d_ws, size_t ws_size,
                              hipStream_t stream) {
  const int FIN = 64, HID = 128, G = 4096;
  const int NH = in_sizes[0] / FIN;
  const int NT = in_sizes[1] / FIN;
  const int EH = in_sizes[20] / 2;
  const int ET = in_sizes[21] / 2;
  const int EB = in_sizes[22] / 2;

  const float* h_x       = (const float*)d_in[0];
  const float* t_x       = (const float*)d_in[1];
  const float* lin0_W    = (const float*)d_in[2];
  const float* lin0_b    = (const float*)d_in[3];
  const float* conv_W    = (const float*)d_in[4];
  const float* conv_asrc = (const float*)d_in[5];
  const float* conv_adst = (const float*)d_in[6];
  const float* conv_b    = (const float*)d_in[7];
  const float* intra_W   = (const float*)d_in[8];
  const float* intra_asrc= (const float*)d_in[9];
  const float* intra_adst= (const float*)d_in[10];
  const float* intra_b   = (const float*)d_in[11];
  const float* inter_Wsrc= (const float*)d_in[12];
  const float* inter_Wdst= (const float*)d_in[13];
  const float* inter_asrc= (const float*)d_in[14];
  const float* inter_adst= (const float*)d_in[15];
  const float* inter_b   = (const float*)d_in[16];
  const float* sag_Wr    = (const float*)d_in[17];
  const float* sag_Wn    = (const float*)d_in[18];
  const float* sag_b     = (const float*)d_in[19];
  const int* h_ei        = (const int*)d_in[20];
  const int* t_ei        = (const int*)d_in[21];
  const int* b_ei        = (const int*)d_in[22];
  const int* h_batch     = (const int*)d_in[23];
  const int* t_batch     = (const int*)d_in[24];

  const int *h_src = h_ei, *h_dst = h_ei + EH;
  const int *t_src = t_ei, *t_dst = t_ei + ET;
  const int *b_src = b_ei, *b_dst = b_ei + EB;

  float* out   = (float*)d_out;
  float* h_rep = out;
  float* t_rep = h_rep + (long long)NH * HID;
  float* h_emb = t_rep + (long long)NT * HID;
  float* t_emb = h_emb + (long long)G * HID;

  float* ws = (float*)d_ws;
  long long wo = 0;
  auto walloc = [&](long long n) { float* p = ws + wo; wo += n; return p; };
  float* Ah = walloc((long long)NH * HID);
  float* At = walloc((long long)NT * HID);
  float* Bh = walloc((long long)NH * HID);
  float* Bt = walloc((long long)NT * HID);
  float* Ch = walloc((long long)NH * HID);
  float* Ct = walloc((long long)NT * HID);
  float* Dh = walloc((long long)NH * HID);
  float* Dt = walloc((long long)NT * HID);
  float* Ehd = walloc((long long)NH * 64);
  float* Etd = walloc((long long)NT * 64);
  float* s0 = walloc((long long)2 * NH);
  float* s1 = walloc((long long)2 * NT);
  float* s2 = walloc((long long)2 * NT);
  float* s3 = walloc((long long)2 * NH);
  float* mB = walloc((long long)2 * (NH > NT ? NH : NT));
  float* dB = walloc((long long)2 * (NH > NT ? NH : NT));
  float* exb = walloc((long long)2 * ((EH > EB ? EH : EB) + NH));

  auto fill = [&](float* p, long long n, float v) {
    fill_kernel<<<(int)((n + 255) / 256), 256, 0, stream>>>(p, n, v);
  };
  auto gemm = [&](const float* X, const float* W, const float* bias, float* Y,
                  int N, int K, int M) {
    int blocks = (N + 63) / 64;
    if (K == 64 && M == 128)
      gemm_bf16_wmma<64, 128><<<blocks, 128, 0, stream>>>(X, W, bias, Y, N);
    else if (K == 128 && M == 128)
      gemm_bf16_wmma<128, 128><<<blocks, 128, 0, stream>>>(X, W, bias, Y, N);
    else
      gemm_bf16_wmma<128, 64><<<blocks, 128, 0, stream>>>(X, W, bias, Y, N);
  };
  auto scores = [&](const float* feat, const float* a, float* o, int N, int H, int F) {
    scores_kernel<<<(N * H + 255) / 256, 256, 0, stream>>>(feat, a, o, N, H, F);
  };
  auto biasact = [&](float* Y, const float* b, int N, int C, int stride, int cofs, int relu) {
    long long tot = (long long)N * C;
    biasact_kernel<<<(int)((tot + 255) / 256), 256, 0, stream>>>(Y, b, N, C, stride, cofs, relu);
  };
  auto gat = [&](const int* src, const int* dst, const float* ssrc, const float* sdst,
                 const float* feat, float* outp, int E, int Ndst, int H, int F,
                 int ostride, int cofs, bool self) {
    int Nself = self ? Ndst : 0;
    fill(mB, (long long)Ndst * H, -INFINITY);
    fill(dB, (long long)Ndst * H, 0.0f);
    int totE = (E + Nself) * H;
    edge_max_kernel<<<(totE + 255) / 256, 256, 0, stream>>>(src, dst, ssrc, sdst, mB, E, Nself, H);
    edge_expsum_kernel<<<(totE + 255) / 256, 256, 0, stream>>>(src, dst, ssrc, sdst, mB, dB, exb, E, Nself, H);
    long long tot = (long long)(E + Nself) * H * F;
    edge_scatter_kernel<<<(int)((tot + 255) / 256), 256, 0, stream>>>(
        src, dst, feat, exb, dB, outp, E, Nself, H, F, ostride, cofs);
  };

  // ---- lin0 ----
  gemm(h_x, lin0_W, lin0_b, Ah, NH, FIN, HID);
  gemm(t_x, lin0_W, lin0_b, At, NT, FIN, HID);
  // ---- shared GATConv (1 head, 128) ----
  gemm(Ah, conv_W, nullptr, Bh, NH, HID, HID);
  gemm(At, conv_W, nullptr, Bt, NT, HID, HID);
  scores(Bh, conv_asrc, s0, NH, 1, HID);
  scores(Bh, conv_adst, s3, NH, 1, HID);
  fill(Ch, (long long)NH * HID, 0.0f);
  gat(h_src, h_dst, s0, s3, Bh, Ch, EH, NH, 1, HID, HID, 0, true);
  biasact(Ch, conv_b, NH, HID, HID, 0, 1);
  scores(Bt, conv_asrc, s1, NT, 1, HID);
  scores(Bt, conv_adst, s2, NT, 1, HID);
  fill(Ct, (long long)NT * HID, 0.0f);
  gat(t_src, t_dst, s1, s2, Bt, Ct, ET, NT, 1, HID, HID, 0, true);
  biasact(Ct, conv_b, NT, HID, HID, 0, 1);
  // ---- elu (shared input of intra & inter) ----
  elu_kernel<<<(int)(((long long)NH * HID + 255) / 256), 256, 0, stream>>>(Ch, Dh, (long long)NH * HID);
  elu_kernel<<<(int)(((long long)NT * HID + 255) / 256), 256, 0, stream>>>(Ct, Dt, (long long)NT * HID);
  fill(h_rep, (long long)NH * HID, 0.0f);
  fill(t_rep, (long long)NT * HID, 0.0f);
  // ---- intra (2 heads x 32) -> rep cols [0,64) ----
  float* hfi = Bh;                         // reuse
  gemm(Dh, intra_W, nullptr, hfi, NH, HID, 64);
  scores(hfi, intra_asrc, s0, NH, 2, 32);
  scores(hfi, intra_adst, s3, NH, 2, 32);
  gat(h_src, h_dst, s0, s3, hfi, h_rep, EH, NH, 2, 32, HID, 0, true);
  biasact(h_rep, intra_b, NH, 64, HID, 0, 0);
  float* tfi = Bt;
  gemm(Dt, intra_W, nullptr, tfi, NT, HID, 64);
  scores(tfi, intra_asrc, s1, NT, 2, 32);
  scores(tfi, intra_adst, s2, NT, 2, 32);
  gat(t_src, t_dst, s1, s2, tfi, t_rep, ET, NT, 2, 32, HID, 0, true);
  biasact(t_rep, intra_b, NT, 64, HID, 0, 0);
  // ---- inter (bipartite, both directions) -> rep cols [64,128) ----
  float* hs = Bh + (long long)NH * 64;
  float* ts = Bt + (long long)NT * 64;
  float* hd = Ehd;
  float* td = Etd;
  gemm(Dh, inter_Wsrc, nullptr, hs, NH, HID, 64);
  gemm(Dh, inter_Wdst, nullptr, hd, NH, HID, 64);
  gemm(Dt, inter_Wsrc, nullptr, ts, NT, HID, 64);
  gemm(Dt, inter_Wdst, nullptr, td, NT, HID, 64);
  scores(hs, inter_asrc, s0, NH, 2, 32);
  scores(td, inter_adst, s1, NT, 2, 32);
  scores(ts, inter_asrc, s2, NT, 2, 32);
  scores(hd, inter_adst, s3, NH, 2, 32);
  // t_inter: edges bs->bd, softmax over t-dst, scatter hs
  gat(b_src, b_dst, s0, s1, hs, t_rep, EB, NT, 2, 32, HID, 64, false);
  biasact(t_rep, inter_b, NT, 64, HID, 64, 0);
  // h_inter: edges bd->bs, softmax over h-dst, scatter ts
  gat(b_dst, b_src, s2, s3, ts, h_rep, EB, NH, 2, 32, HID, 64, false);
  biasact(h_rep, inter_b, NH, 64, HID, 64, 0);
  // ---- SAG pool (h) ----
  fill(Ah, (long long)NH * HID, 0.0f);
  sag_agg_kernel<<<(int)(((long long)EH * 128 + 255) / 256), 256, 0, stream>>>(h_src, h_dst, h_rep, Ah, EH);
  sag_score_kernel<<<(NH + 255) / 256, 256, 0, stream>>>(h_rep, Ah, sag_Wr, sag_Wn, sag_b, s0, NH);
  fill(mB, G, -INFINITY);
  fill(dB, G, 0.0f);
  seg_max_kernel<<<(NH + 255) / 256, 256, 0, stream>>>(s0, h_batch, mB, NH);
  seg_expsum_kernel<<<(NH + 255) / 256, 256, 0, stream>>>(s0, h_batch, mB, dB, s3, NH);
  fill(h_emb, (long long)G * HID, 0.0f);
  sag_pool_kernel<<<(int)(((long long)NH * 128 + 255) / 256), 256, 0, stream>>>(h_rep, h_batch, s3, dB, h_emb, NH);
  // ---- SAG pool (t) ----
  fill(At, (long long)NT * HID, 0.0f);
  sag_agg_kernel<<<(int)(((long long)ET * 128 + 255) / 256), 256, 0, stream>>>(t_src, t_dst, t_rep, At, ET);
  sag_score_kernel<<<(NT + 255) / 256, 256, 0, stream>>>(t_rep, At, sag_Wr, sag_Wn, sag_b, s1, NT);
  fill(mB, G, -INFINITY);
  fill(dB, G, 0.0f);
  seg_max_kernel<<<(NT + 255) / 256, 256, 0, stream>>>(s1, t_batch, mB, NT);
  seg_expsum_kernel<<<(NT + 255) / 256, 256, 0, stream>>>(s1, t_batch, mB, dB, s2, NT);
  fill(t_emb, (long long)G * HID, 0.0f);
  sag_pool_kernel<<<(int)(((long long)NT * 128 + 255) / 256), 256, 0, stream>>>(t_rep, t_batch, s2, dB, t_emb, NT);

  (void)n_in; (void)out_size; (void)ws_size;
}
